// Block_69054484185387
// MI455X (gfx1250) — matrix operand order repaired and checked by reading, not compiled, not run on hardware
//
#include <hip/hip_runtime.h>
#include <hip/hip_bf16.h>
#include <math.h>

typedef __attribute__((ext_vector_type(16))) _Float16 v16h;
typedef __attribute__((ext_vector_type(8)))  _Float16 v8h;
typedef __attribute__((ext_vector_type(8)))  float    v8f;

#define WMMA_F32_F16(a, b, c) \
  __builtin_amdgcn_wmma_f32_16x16x32_f16(false, (a), false, (b), (short)0, (c), false, false)

#define CAT16(lo, hi)                                         \
  __builtin_shufflevector((lo), (hi), 0, 1, 2, 3, 4, 5, 6, 7, \
                          8, 9, 10, 11, 12, 13, 14, 15)

// ds_swizzle group-of-32 XOR butterfly: offset = and_mask(0x1f) | xor<<10
#define DSWZ(x, m) __builtin_amdgcn_ds_swizzle((x), (0x1f | ((m) << 10)))

__device__ __forceinline__ float red16_max(float v) {
  v = fmaxf(v, __int_as_float(DSWZ(__float_as_int(v), 1)));
  v = fmaxf(v, __int_as_float(DSWZ(__float_as_int(v), 2)));
  v = fmaxf(v, __int_as_float(DSWZ(__float_as_int(v), 4)));
  v = fmaxf(v, __int_as_float(DSWZ(__float_as_int(v), 8)));
  return v;
}
__device__ __forceinline__ float red16_sum(float v) {
  v += __int_as_float(DSWZ(__float_as_int(v), 1));
  v += __int_as_float(DSWZ(__float_as_int(v), 2));
  v += __int_as_float(DSWZ(__float_as_int(v), 4));
  v += __int_as_float(DSWZ(__float_as_int(v), 8));
  return v;
}

// Branchless exact-GELU: erf via Abramowitz-Stegun 7.1.26 (|err|<=1.5e-7).
__device__ __forceinline__ float gelu_exact(float x) {
  const float ax = fabsf(x) * 0.70710678118654752f;
  const float t = 1.0f / (1.0f + 0.3275911f * ax);
  const float poly =
      t * (0.254829592f +
           t * (-0.284496736f +
                t * (1.421413741f + t * (-1.453152027f + t * 1.061405429f))));
  const float erf_ax = 1.0f - poly * __expf(-ax * ax);
  return 0.5f * x * (1.0f + copysignf(erf_ax, x));
}

#define PACK4(dst, base, f4)                     \
  (dst)[(base) + 0] = (_Float16)(f4).x;          \
  (dst)[(base) + 1] = (_Float16)(f4).y;          \
  (dst)[(base) + 2] = (_Float16)(f4).z;          \
  (dst)[(base) + 3] = (_Float16)(f4).w;

// ---------------------------------------------------------------------------
// LayerNorm: one block (256 threads) per row of D elements.
// ---------------------------------------------------------------------------
__global__ __launch_bounds__(256) void ln_kernel(
    const float* __restrict__ x, const float* __restrict__ g,
    const float* __restrict__ be, float* __restrict__ out, int D) {
  __shared__ float red[256];
  const long row = blockIdx.x;
  const float* xr = x + row * (long)D;
  const int tid = threadIdx.x;

  float s = 0.f;
  for (int i = tid; i < D; i += 256) s += xr[i];
  red[tid] = s;
  __syncthreads();
  for (int off = 128; off > 0; off >>= 1) {
    if (tid < off) red[tid] += red[tid + off];
    __syncthreads();
  }
  const float mu = red[0] / (float)D;
  __syncthreads();

  float s2 = 0.f;
  for (int i = tid; i < D; i += 256) {
    const float d = xr[i] - mu;
    s2 += d * d;
  }
  red[tid] = s2;
  __syncthreads();
  for (int off = 128; off > 0; off >>= 1) {
    if (tid < off) red[tid] += red[tid + off];
    __syncthreads();
  }
  const float rstd = rsqrtf(red[0] / (float)D + 1e-5f);

  for (int i = tid; i < D; i += 256)
    out[row * (long)D + i] = (xr[i] - mu) * rstd * g[i] + be[i];
}

// ---------------------------------------------------------------------------
// Generic WMMA GEMM: out[z] = act(A[z] * B[z] + bias) + resid
// A: row-major MxK fp32. B: row-major KxN fp32. Block tile 256x64, BK=32,
// 8 waves in a 4x2 grid; each wave owns 64x32 (4x2 WMMA tiles -> 8 WMMAs
// per 12 ds_load_b128). Register double-buffered staging. Template params
// fold all epilogue control flow at compile time.
// CMODE: 0 = fp32 C, 1 = f16 Ch, 2 = f16 Ch transposed (Ch[n*ldc+m]).
// Batched via grid.z: b=z/HH -> strideAb, h=z%HH -> strideBh, z -> strideCz.
// ---------------------------------------------------------------------------
#define BM 256
#define BN 64
#define BK 32
#define KP 40  // padded K stride in halves: 80B, 16B-aligned, conflict-light

template <int CMODE, bool ACT, bool BIAS, bool RESID>
__global__ __launch_bounds__(256) void gemm_wmma_kernel(
    const float* __restrict__ A, const float* __restrict__ Bm,
    float* __restrict__ C, _Float16* __restrict__ Ch,
    const float* __restrict__ bias, const float* __restrict__ resid, int N,
    int K, int ldc, int HH, long strideAb, long strideBh, long strideCz) {
  const int z = blockIdx.z;
  A += (long)(z / HH) * strideAb;
  Bm += (long)(z % HH) * strideBh;
  if (CMODE == 0) C += (long)z * strideCz;
  else            Ch += (long)z * strideCz;

  __shared__ alignas(16) _Float16 As[BM][KP];  // row-major
  __shared__ alignas(16) _Float16 Bs[BN][KP];  // col-major (k contiguous)

  const int tid = threadIdx.x;
  const int wid = tid >> 5, lane = tid & 31;
  const int wm = wid & 3, wn = wid >> 2;  // 4x2 wave grid
  const int ln = lane & 15, lh = lane >> 4;
  const int row0 = blockIdx.x * BM;
  const int col0 = blockIdx.y * BN;

  const int colB = tid & 63, kbB = (tid >> 6) << 3;  // B staging mapping

  v8f acc[4][2] = {};
  float4 ar[8];
  float br[8];

  // ---- prologue: stage k0 = 0 ----
#pragma unroll
  for (int i = 0; i < 8; ++i) {
    const int idx = tid + i * 256;
    const int r = idx >> 3, kc = (idx & 7) << 2;
    ar[i] = *reinterpret_cast<const float4*>(A + (long)(row0 + r) * K + kc);
  }
#pragma unroll
  for (int j = 0; j < 8; ++j)
    br[j] = Bm[(long)(kbB + j) * N + col0 + colB];

#pragma unroll
  for (int i = 0; i < 8; ++i) {
    const int idx = tid + i * 256;
    const int r = idx >> 3, kc = (idx & 7) << 2;
    PACK4(As[r], kc, ar[i]);
  }
  {
    v8h b8;
#pragma unroll
    for (int j = 0; j < 8; ++j) b8[j] = (_Float16)br[j];
    *reinterpret_cast<v8h*>(&Bs[colB][kbB]) = b8;
  }
  __syncthreads();

  for (int k0 = 0; k0 < K; k0 += BK) {
    const bool has_next = (k0 + BK) < K;
    // ---- issue next tile's global loads (latency hidden behind WMMAs) ----
    if (has_next) {
      const int kn = k0 + BK;
#pragma unroll
      for (int i = 0; i < 8; ++i) {
        const int idx = tid + i * 256;
        const int r = idx >> 3, kc = (idx & 7) << 2;
        ar[i] =
            *reinterpret_cast<const float4*>(A + (long)(row0 + r) * K + kn + kc);
      }
#pragma unroll
      for (int j = 0; j < 8; ++j)
        br[j] = Bm[(long)(kn + kbB + j) * N + col0 + colB];
    }

    // ---- fragments (2x ds_load_b128 each) + 8 WMMAs ----
    v16h bf[2];
#pragma unroll
    for (int t = 0; t < 2; ++t) {
      const int bcol = wn * 32 + t * 16 + ln;
      const v8h b0 = *reinterpret_cast<const v8h*>(&Bs[bcol][lh * 8]);
      const v8h b1 = *reinterpret_cast<const v8h*>(&Bs[bcol][16 + lh * 8]);
      bf[t] = CAT16(b0, b1);
    }
#pragma unroll
    for (int tm = 0; tm < 4; ++tm) {
      const int arow = wm * 64 + tm * 16 + ln;
      const v8h a0 = *reinterpret_cast<const v8h*>(&As[arow][lh * 8]);
      const v8h a1 = *reinterpret_cast<const v8h*>(&As[arow][16 + lh * 8]);
      const v16h af = CAT16(a0, a1);
      acc[tm][0] = WMMA_F32_F16(af, bf[0], acc[tm][0]);
      acc[tm][1] = WMMA_F32_F16(af, bf[1], acc[tm][1]);
    }
    __syncthreads();

    if (has_next) {
#pragma unroll
      for (int i = 0; i < 8; ++i) {
        const int idx = tid + i * 256;
        const int r = idx >> 3, kc = (idx & 7) << 2;
        PACK4(As[r], kc, ar[i]);
      }
      v8h b8;
#pragma unroll
      for (int j = 0; j < 8; ++j) b8[j] = (_Float16)br[j];
      *reinterpret_cast<v8h*>(&Bs[colB][kbB]) = b8;
      __syncthreads();
    }
  }

  // ---- epilogue: bias -> exact GELU -> residual (all folded by template) ----
#pragma unroll
  for (int tn = 0; tn < 2; ++tn) {
    const int col = col0 + wn * 32 + tn * 16 + ln;
    const float bv = BIAS ? bias[col] : 0.f;
#pragma unroll
    for (int tm = 0; tm < 4; ++tm) {
#pragma unroll
      for (int r = 0; r < 8; ++r) {
        const int row = row0 + wm * 64 + tm * 16 + r + lh * 8;
        float v = acc[tm][tn][r] + bv;
        if (ACT) v = gelu_exact(v);
        if (RESID) v += resid[(long)row * ldc + col];
        if (CMODE == 0)      C[(long)row * ldc + col] = v;
        else if (CMODE == 1) Ch[(long)row * ldc + col] = (_Float16)v;
        else                 Ch[(long)col * ldc + row] = (_Float16)v;
      }
    }
  }
}

// ---------------------------------------------------------------------------
// Flash attention (causal). q,k: f16 [(b*H+h)*P + p]*64 + e. vT: f16
// [(b*H+h)*64 + e]*P + p (transposed by the V-projection GEMM epilogue).
// Block = 8 waves; wave w owns query rows [q0, q0+16). 32 keys per step.
// Q is pre-scaled by D^-0.5 (=1/32, exact in f16), so no per-step scaling.
// The causal mask is applied only on the single diagonal block per wave
// (MASKED=true instantiation); all earlier blocks take the maskless path.
// ---------------------------------------------------------------------------
template <bool MASKED>
__device__ __forceinline__ void flash_kv_step(
    const int kv0, const int q0, const int ln, const int lh, const int P,
    const _Float16* __restrict__ kb, const _Float16* __restrict__ vb,
    const v16h (&qa)[2], float (&mi)[8], float (&li)[8], v8f (&acc)[4],
    _Float16 (&psw)[16][KP]) {
  if (!MASKED) {  // next block always exists on the maskless path
    __builtin_prefetch(kb + (long)(kv0 + 32 + ln + lh * 16) * 64, 0, 1);
    __builtin_prefetch(vb + (long)(ln + lh * 16) * P + kv0 + 32, 0, 1);
  }

  // ---- S = Q K^T (Q pre-scaled) ----
  v8f st[2];
#pragma unroll
  for (int s = 0; s < 2; ++s) {
    const _Float16* ksrc = kb + (long)(kv0 + s * 16 + ln) * 64;
    v16h kf[2];
#pragma unroll
    for (int c = 0; c < 2; ++c) {
      const _Float16* p0 = ksrc + c * 32 + lh * 8;
      const v8h r0 = *reinterpret_cast<const v8h*>(p0);
      const v8h r1 = *reinterpret_cast<const v8h*>(p0 + 16);
      kf[c] = CAT16(r0, r1);
    }
    v8f zacc = {};
    zacc = WMMA_F32_F16(qa[0], kf[0], zacc);
    st[s] = WMMA_F32_F16(qa[1], kf[1], zacc);
  }

  // ---- causal mask: diagonal block only ----
  if (MASKED) {
#pragma unroll
    for (int s = 0; s < 2; ++s) {
#pragma unroll
      for (int r = 0; r < 8; ++r) {
        const int row = q0 + r + lh * 8;
        const int col = kv0 + s * 16 + ln;
        st[s][r] = (col <= row) ? st[s][r] : -INFINITY;
      }
    }
  }

  // ---- online softmax ----
#pragma unroll
  for (int r = 0; r < 8; ++r) {
    const float vm = red16_max(fmaxf(st[0][r], st[1][r]));
    const float mn = fmaxf(mi[r], vm);
    const float alpha = __expf(mi[r] - mn);  // 0 on first block
    mi[r] = mn;
    st[0][r] = __expf(st[0][r] - mn);
    st[1][r] = __expf(st[1][r] - mn);
    const float rs = red16_sum(st[0][r] + st[1][r]);
    li[r] = li[r] * alpha + rs;
#pragma unroll
    for (int t = 0; t < 4; ++t) acc[t][r] *= alpha;
  }

  // ---- transpose P (D-layout -> A-layout) via per-wave LDS ----
#pragma unroll
  for (int s = 0; s < 2; ++s)
#pragma unroll
    for (int r = 0; r < 8; ++r)
      psw[r + lh * 8][s * 16 + ln] = (_Float16)st[s][r];
  asm volatile("s_wait_dscnt 0" ::: "memory");
  const v8h p0 = *reinterpret_cast<const v8h*>(&psw[ln][lh * 8]);
  const v8h p1 = *reinterpret_cast<const v8h*>(&psw[ln][16 + lh * 8]);
  const v16h pa = CAT16(p0, p1);

  // ---- O += P * V : V^T rows give contiguous key runs per lane ----
#pragma unroll
  for (int t = 0; t < 4; ++t) {
    const _Float16* vsrc = vb + (long)(t * 16 + ln) * P + kv0 + lh * 8;
    const v8h r0 = *reinterpret_cast<const v8h*>(vsrc);
    const v8h r1 = *reinterpret_cast<const v8h*>(vsrc + 16);
    const v16h vf = CAT16(r0, r1);
    acc[t] = WMMA_F32_F16(pa, vf, acc[t]);
  }
}

__global__ __launch_bounds__(256) void flash_attn_kernel(
    const _Float16* __restrict__ qm, const _Float16* __restrict__ km,
    const _Float16* __restrict__ vtm, float* __restrict__ o, int P, int H,
    float scale) {
  __shared__ alignas(16) _Float16 ps[8][16][KP];  // per-wave P transpose

  const int bh = blockIdx.x;
  const int b = bh / H, hh = bh % H;
  const int tid = threadIdx.x;
  const int wid = tid >> 5, lane = tid & 31;
  const int ln = lane & 15, lh = lane >> 4;
  const int q0 = blockIdx.y * 128 + wid * 16;

  const _Float16* qb = qm + (long)bh * P * 64;
  const _Float16* kb = km + (long)bh * P * 64;
  const _Float16* vb = vtm + (long)bh * 64 * P;

  // Q fragments (register resident), pre-scaled by D^-0.5 (exact f16 pow2).
  const _Float16 hs = (_Float16)scale;
  v16h qa[2];
#pragma unroll
  for (int c = 0; c < 2; ++c) {
    const _Float16* src = qb + (long)(q0 + ln) * 64 + c * 32 + lh * 8;
    const v8h r0 = *reinterpret_cast<const v8h*>(src);
    const v8h r1 = *reinterpret_cast<const v8h*>(src + 16);
    qa[c] = CAT16(r0, r1) * hs;
  }

  float mi[8], li[8];
#pragma unroll
  for (int r = 0; r < 8; ++r) { mi[r] = -INFINITY; li[r] = 0.f; }
  v8f acc[4] = {};

  // Full (maskless) blocks, then the single masked diagonal block.
  const int diag = q0 & ~31;
  for (int kv0 = 0; kv0 < diag; kv0 += 32)
    flash_kv_step<false>(kv0, q0, ln, lh, P, kb, vb, qa, mi, li, acc, ps[wid]);
  flash_kv_step<true>(diag, q0, ln, lh, P, kb, vb, qa, mi, li, acc, ps[wid]);

  // ---- normalize + store concat-heads (fp32) ----
  const int D = H * 64;
#pragma unroll
  for (int t = 0; t < 4; ++t)
#pragma unroll
    for (int r = 0; r < 8; ++r) {
      const int row = q0 + r + lh * 8;
      o[((long)b * P + row) * D + hh * 64 + t * 16 + ln] = acc[t][r] / li[r];
    }
}

// ---------------------------------------------------------------------------
// Host launcher
// ---------------------------------------------------------------------------
extern "C" void kernel_launch(void* const* d_in, const int* in_sizes, int n_in,
                              void* d_out, int out_size, void* d_ws,
                              size_t ws_size, hipStream_t stream) {
  (void)in_sizes; (void)n_in; (void)out_size; (void)ws_size;
  const int B = 2, P = 2048, D = 1024, H = 16, DH = 64;

  const float* x   = (const float*)d_in[0];
  const float* Wq  = (const float*)d_in[1];
  const float* Wk  = (const float*)d_in[2];
  const float* Wv  = (const float*)d_in[3];
  const float* Wp  = (const float*)d_in[4];
  const float* bp  = (const float*)d_in[5];
  const float* W1  = (const float*)d_in[6];
  const float* b1  = (const float*)d_in[7];
  const float* W2  = (const float*)d_in[8];
  const float* b2  = (const float*)d_in[9];
  const float* g1  = (const float*)d_in[10];
  const float* be1 = (const float*)d_in[11];
  const float* g2  = (const float*)d_in[12];
  const float* be2 = (const float*)d_in[13];
  float* out = (float*)d_out;

  const long npd = (long)B * P * D;  // 4M
  float* ws  = (float*)d_ws;
  float* hln = ws;             // [B*P, D]       (reused as h2)
  float* x1  = ws + 1 * npd;   // [B*P, D]
  // f16 region: q/k/vT, 4M halves (2M floats) each, starting at ws+8M
  _Float16* qh = (_Float16*)(ws + 2 * npd);
  _Float16* kh = qh + npd;
  _Float16* vh = kh + npd;     // transposed: [B*H][64][P]
  float* attn = ws + 2 * npd + 3 * (npd / 2);  // ws+14M, [B*P, D] fp32
  float* mid  = ws + 2 * npd;  // [B*P, 4D] aliases q/k/vT/attn (dead by then)

  const dim3 blk(256);
  const int BP = B * P;

  // 1) LN1
  ln_kernel<<<BP, blk, 0, stream>>>(x, g1, be1, hln, D);

  // 2) QKV: per-(b,h) batched GEMM, M=P, N=DH, K=D; f16 outputs, V transposed
  {
    const dim3 grid(P / BM, 1, B * H);
    const long sA = (long)P * D, sB = (long)D * DH, sC = (long)P * DH;
    gemm_wmma_kernel<1, false, false, false><<<grid, blk, 0, stream>>>(
        hln, Wq, nullptr, qh, nullptr, nullptr, DH, D, DH, H, sA, sB, sC);
    gemm_wmma_kernel<1, false, false, false><<<grid, blk, 0, stream>>>(
        hln, Wk, nullptr, kh, nullptr, nullptr, DH, D, DH, H, sA, sB, sC);
    gemm_wmma_kernel<2, false, false, false><<<grid, blk, 0, stream>>>(
        hln, Wv, nullptr, vh, nullptr, nullptr, DH, D, P, H, sA, sB, sC);
  }

  // 3) causal flash attention, scale = D^-0.5 = 1/32
  flash_attn_kernel<<<dim3(B * H, P / 128), blk, 0, stream>>>(
      qh, kh, vh, attn, P, H, 1.0f / 32.0f);

  // 4) x1 = x + attn @ Wp + bp
  gemm_wmma_kernel<0, false, true, true><<<dim3(BP / BM, D / BN, 1), blk, 0,
                                           stream>>>(
      attn, Wp, x1, nullptr, bp, x, D, D, D, 1, 0, 0, 0);

  // 5) LN2 (reuse hln as h2)
  ln_kernel<<<BP, blk, 0, stream>>>(x1, g2, be2, hln, D);

  // 6) mid = gelu(h2 @ W1 + b1)
  gemm_wmma_kernel<0, true, true, false><<<dim3(BP / BM, (4 * D) / BN, 1), blk,
                                           0, stream>>>(
      hln, W1, mid, nullptr, b1, nullptr, 4 * D, D, 4 * D, 1, 0, 0, 0);

  // 7) out = x1 + mid @ W2 + b2
  gemm_wmma_kernel<0, false, true, true><<<dim3(BP / BM, D / BN, 1), blk, 0,
                                           stream>>>(
      mid, W2, out, nullptr, b2, x1, D, 4 * D, D, 1, 0, 0, 0);
}
